// BoundaryDecision_62294205661620
// MI455X (gfx1250) — matrix operand order repaired and checked by reading, not compiled, not run on hardware
//
#include <hip/hip_runtime.h>
#include <hip/hip_bf16.h>
#include <math.h>

// ---------------- problem constants ----------------
constexpr int kHID  = 1024;
constexpr int kNH   = 16;
constexpr int kHD   = 64;
constexpr int kBS   = 4;
constexpr int kQS   = 2048;
constexpr int kROWS = kBS * kQS;       // 8192
constexpr int kQKN  = 2 * kNH * kHD;   // 2048 (Q cols 0..1023, K cols 1024..2047)
constexpr float kNEG   = -60000.0f;
constexpr float kSCALE = 0.125f;       // 1/sqrt(64)

// ---------------- CDNA5 WMMA types ----------------
typedef __attribute__((ext_vector_type(16))) __bf16         v16bf;
typedef __attribute__((ext_vector_type(8)))  float          v8f;
typedef __attribute__((ext_vector_type(16))) unsigned short v16u;
typedef __attribute__((ext_vector_type(8)))  unsigned short v8u;

__device__ __forceinline__ unsigned short f2bf(float f) {
  unsigned int u = __float_as_uint(f);
  u += 0x7fffu + ((u >> 16) & 1u);      // round-to-nearest-even
  return (unsigned short)(u >> 16);
}

// 16 contiguous bf16 (B-operand: element i <-> K = i + 16*h)
__device__ __forceinline__ v16bf load16(const unsigned short* p) {
  union { v16u u; v16bf b; } t;
  t.u = *(const v16u*)p;
  return t.b;
}

// A-operand: element i <-> K = i + 16*h : two 8-element runs
__device__ __forceinline__ v16bf make_frag(v8u lo, v8u hi) {
  union { v16u u; v16bf b; } t;
#pragma unroll
  for (int i = 0; i < 8; ++i) { t.u[i] = lo[i]; t.u[i + 8] = hi[i]; }
  return t.b;
}

__device__ __forceinline__ v8f wmma_bf16(v16bf a, v16bf b, v8f c) {
  return __builtin_amdgcn_wmma_f32_16x16x32_bf16(
      /*neg_a=*/false, a, /*neg_b=*/false, b,
      /*c_mod=*/(short)0, c, /*reuse_a=*/false, /*reuse_b=*/false);
}

// ---------------- Tensor Data Mover (TDM) helper ----------------
#if __has_builtin(__builtin_amdgcn_tensor_load_to_lds) && \
    __has_builtin(__builtin_amdgcn_s_wait_tensorcnt)
#define HAS_TDM 1
typedef __attribute__((ext_vector_type(4))) unsigned int u32x4;
typedef __attribute__((ext_vector_type(8))) int          i32x8;
typedef __attribute__((ext_vector_type(4))) int          i32x4;

// One 1-D contiguous tile: nelem f32 from gsrc -> LDS[lds_off].
// D# bit layout per CDNA5 ISA ch.8: group0 = {count/lds_addr/global_addr/type},
// group1 = {data_size, tensor/tile dims, strides}; groups 2/3 unused (zero).
// This toolchain exposes the 6-arg builtin (extra int32x8 group slot + cpol).
__device__ __forceinline__ void tdm_row_to_lds(const float* gsrc,
                                               unsigned lds_off, int nelem) {
  unsigned long long ga = (unsigned long long)(size_t)gsrc;
  u32x4 g0;
  g0[0] = 1u;                                        // count=1, user descriptor
  g0[1] = lds_off;                                   // lds_addr (bytes)
  g0[2] = (unsigned)(ga & 0xffffffffu);              // global_addr[31:0]
  g0[3] = (unsigned)((ga >> 32) & 0x01ffffffu)       // global_addr[56:32]
          | 0x80000000u;                             // type = 2 ("image")
  i32x8 g1;
  g1[0] = 2 << 16;                                   // data_size = 4B
  g1[1] = (nelem & 0xffff) << 16;                    // tensor_dim0[15:0]
  g1[2] = ((nelem >> 16) & 0xffff) | (1 << 16);      // dim0[31:16], tensor_dim1=1
  g1[3] = (nelem & 0xffff) << 16;                    // tile_dim0
  g1[4] = 0;                                         // tile_dim1/2 unused
  g1[5] = nelem;                                     // tensor_dim0_stride[31:0]
  g1[6] = 0;
  g1[7] = 0;
  i32x4 z4 = {0, 0, 0, 0};
  i32x8 z8 = {0, 0, 0, 0, 0, 0, 0, 0};
  __builtin_amdgcn_tensor_load_to_lds(g0, g1, z4, z4, z8, 0);
}
#endif

// ---------------- prep kernels ----------------
__global__ void k_cast_bf16(const float* __restrict__ src,
                            unsigned short* __restrict__ dst, int n) {
  int i = blockIdx.x * blockDim.x + threadIdx.x;
  if (i < n) dst[i] = f2bf(src[i]);
}

// Wt[n][k] = bf16(W_qk[k][n])  (column-major bf16 so B-fragments are contiguous)
__global__ void k_w_transpose(const float* __restrict__ W,
                              unsigned short* __restrict__ Wt) {
  int i = blockIdx.x * blockDim.x + threadIdx.x;   // n*kHID + k
  int n = i >> 10, k = i & (kHID - 1);
  Wt[i] = f2bf(W[(size_t)k * kQKN + n]);
}

// ---------------- qk = x @ W_qk via WMMA ----------------
// grid (kROWS/16, kQKN/128), 256 threads: each wave owns one 16x16 tile,
// K-loop of 32 bf16 chunks (32 wmma per wave). A/B use k = 16h + i order; C per ISA table.
__global__ void __launch_bounds__(256)
k_gemm_qk(const unsigned short* __restrict__ xb,
          const unsigned short* __restrict__ wt,
          unsigned short* __restrict__ qk) {
  const int lane = threadIdx.x & 31;
  const int wave = threadIdx.x >> 5;
  const int l16  = lane & 15;
  const int h    = lane >> 4;
  const int mbase = blockIdx.x * 16;
  const int nbase = (blockIdx.y * 8 + wave) * 16;

  const unsigned short* arow = xb + (size_t)(mbase + l16) * kHID;
  const unsigned short* brow = wt + (size_t)(nbase + l16) * kHID;

  v8f acc = {0.f, 0.f, 0.f, 0.f, 0.f, 0.f, 0.f, 0.f};
#pragma unroll 4
  for (int kc = 0; kc < kHID; kc += 32) {
    __builtin_prefetch(arow + kc + 128, 0, 3);   // global_prefetch_b8, 4 chunks ahead
    __builtin_prefetch(brow + kc + 128, 0, 3);
    v16bf a = make_frag(*(const v8u*)(arow + kc + 16 * h),
                        *(const v8u*)(arow + kc + 8 + 16 * h));
    v16bf b = load16(brow + kc + 16 * h);
    acc = wmma_bf16(a, b, acc);
  }
  // C layout: lane holds column n = nbase+l16; VGPR r holds row mbase + r + 8h
#pragma unroll
  for (int r = 0; r < 8; ++r) {
    int m = mbase + r + 8 * h;
    qk[(size_t)m * kQKN + nbase + l16] = f2bf(acc[r]);
  }
}

// ---------------- vT[head][row] = (x @ W_v)^T  (tiny: N=16, f32 VALU) --------
// Block = 16 rows x 16 heads; LDS mini-transpose so the vT stores (and later
// the per-(b,head) value-column reads / TDM copy) are contiguous.
__global__ void __launch_bounds__(256)
k_value(const float* __restrict__ x, const float* __restrict__ Wv,
        float* __restrict__ vT) {
  __shared__ float tile[16][17];
  const int rowl = threadIdx.x >> 4;   // 0..15
  const int head = threadIdx.x & 15;
  const int row  = blockIdx.x * 16 + rowl;
  const float* xr = x + (size_t)row * kHID;
  float s = 0.f;
#pragma unroll 8
  for (int k = 0; k < kHID; ++k) s += xr[k] * Wv[k * kNH + head];
  tile[head][rowl] = s;
  __syncthreads();
  const int oh = threadIdx.x >> 4;     // head for the write phase
  const int orl = threadIdx.x & 15;    // row-in-block: contiguous 16-float runs
  vT[(size_t)oh * kROWS + blockIdx.x * 16 + orl] = tile[oh][orl];
}

__global__ void k_zero(float* __restrict__ p, int n) {
  int i = blockIdx.x * blockDim.x + threadIdx.x;
  if (i < n) p[i] = 0.f;
}

// ---------------- fused attention ----------------
// block = (qtile of 16 queries, head, batch); 256 threads = 8 waves.
// Wave w owns keys [w*256, w*256+256): 16 K-tiles, scores C[m=kpos][n=q]
// stay in 128 VGPRs (v8f c[16]). The value column is DMA'd into LDS by the
// TDM concurrently with the WMMA phase (TENSORcnt waited before the barrier).
__global__ void __launch_bounds__(256)
k_attn(const unsigned short* __restrict__ qk, const float* __restrict__ vT,
       const unsigned char* __restrict__ mask, float* __restrict__ att_sum) {
  __shared__ float vsh[kQS];        // value column for this (b, head)
  __shared__ float redM[8][16];
  __shared__ float redE[8][16];
  __shared__ float redW[8][16];

  const int qt = blockIdx.x, head = blockIdx.y, b = blockIdx.z;
  const int qbase = qt * 16;
  const int lane = threadIdx.x & 31;
  const int wave = threadIdx.x >> 5;
  const int l16  = lane & 15;
  const int h    = lane >> 4;
  const int kw   = wave * 256;

  const float* vcol = vT + (size_t)head * kROWS + b * kQS;
#ifdef HAS_TDM
  if (wave == 0)                        // one TDM op per workgroup, async
    tdm_row_to_lds(vcol, (unsigned)(size_t)(void*)vsh, kQS);
#else
  for (int i = threadIdx.x; i < kQS; i += 256) vsh[i] = vcol[i];
#endif

  // Q tile as B-operand: lane column n = query qbase+l16, d = i + 16h (+32*kk)
  const unsigned short* qrow =
      qk + (size_t)(b * kQS + qbase + l16) * kQKN + head * kHD;
  const v16bf bq0 = load16(qrow + 16 * h);
  const v16bf bq1 = load16(qrow + 32 + 16 * h);

  const size_t mrow = ((size_t)b * kQS + (qbase + l16)) * (size_t)kQS;

  v8f c[16];
#pragma unroll
  for (int t = 0; t < 16; ++t)
    c[t] = (v8f){0.f, 0.f, 0.f, 0.f, 0.f, 0.f, 0.f, 0.f};

#pragma unroll
  for (int t = 0; t < 16; ++t) {
    const int kb = kw + t * 16;
    // K tile as A-operand: row m = kpos kb+l16, d = i + 16h (+32*kk)
    const unsigned short* krow =
        qk + (size_t)(b * kQS + kb + l16) * kQKN + kNH * kHD + head * kHD;
    if (t < 15) {                        // prefetch next tile's K row + mask
      __builtin_prefetch(krow + kQKN * 16, 0, 3);
      __builtin_prefetch(mask + mrow + kb + 16, 0, 3);
    }
    v16bf a0 = make_frag(*(const v8u*)(krow + 16 * h),
                         *(const v8u*)(krow + 8 + 16 * h));
    v16bf a1 = make_frag(*(const v8u*)(krow + 32 + 16 * h),
                         *(const v8u*)(krow + 40 + 16 * h));
    c[t] = wmma_bf16(a0, bq0, c[t]);
    c[t] = wmma_bf16(a1, bq1, c[t]);
    // mask + scale: lane's 8 kpos = kb + 8h + r  -> one 8-byte bool load
    uint2 mv = *(const uint2*)(mask + mrow + kb + 8 * h);
#pragma unroll
    for (int r = 0; r < 8; ++r) {
      unsigned byte =
          (r < 4 ? (mv.x >> (8 * r)) : (mv.y >> (8 * (r - 4)))) & 0xffu;
      c[t][r] = byte ? kNEG : c[t][r] * kSCALE;
    }
  }

  // ---- softmax phase 1: per-query max over all 2048 keys ----
  float m = -3.0e38f;
#pragma unroll
  for (int t = 0; t < 16; ++t)
#pragma unroll
    for (int r = 0; r < 8; ++r) m = fmaxf(m, c[t][r]);
  m = fmaxf(m, __shfl_xor(m, 16, 32));          // merge lane halves (same q)
  if (h == 0) redM[wave][l16] = m;
#ifdef HAS_TDM
  if (wave == 0) __builtin_amdgcn_s_wait_tensorcnt(0);  // vsh DMA done
#endif
  __syncthreads();
  float gm = -3.0e38f;
#pragma unroll
  for (int w = 0; w < 8; ++w) gm = fmaxf(gm, redM[w][l16]);

  // ---- softmax phase 2: exp, denom, value-weighted numerator ----
  float es = 0.f, ws = 0.f;
#pragma unroll
  for (int t = 0; t < 16; ++t) {
    const float* vp = &vsh[kw + t * 16 + 8 * h];
#pragma unroll
    for (int r = 0; r < 8; ++r) {
      float e = __expf(c[t][r] - gm);
      es += e;
      ws += e * vp[r];
    }
  }
  es += __shfl_xor(es, 16, 32);
  ws += __shfl_xor(ws, 16, 32);
  if (h == 0) { redE[wave][l16] = es; redW[wave][l16] = ws; }
  __syncthreads();
  if (threadIdx.x < 16) {
    float te = 0.f, tw = 0.f;
#pragma unroll
    for (int w = 0; w < 8; ++w) { te += redE[w][threadIdx.x]; tw += redW[w][threadIdx.x]; }
    atomicAdd(&att_sum[b * kQS + qbase + threadIdx.x], tw / te);
  }
}

// ---------------- finalize: row-mask + sigmoid ----------------
__global__ void k_final(const float* __restrict__ att_sum,
                        const unsigned char* __restrict__ mask,
                        float* __restrict__ out) {
  int i = blockIdx.x * blockDim.x + threadIdx.x;   // 8192
  int b = i >> 11, q = i & (kQS - 1);
  float v = mask[((size_t)b * kQS + (kQS - 1)) * kQS + q] ? kNEG : att_sum[i];
  out[i] = 1.f / (1.f + __expf(-v));
}

// ---------------- launch ----------------
extern "C" void kernel_launch(void* const* d_in, const int* in_sizes, int n_in,
                              void* d_out, int out_size, void* d_ws, size_t ws_size,
                              hipStream_t stream) {
  (void)in_sizes; (void)n_in; (void)out_size; (void)ws_size;
  const float*         x    = (const float*)d_in[0];
  const unsigned char* mask = (const unsigned char*)d_in[1];  // numpy bool bytes
  const float*         Wqk  = (const float*)d_in[2];
  const float*         Wv   = (const float*)d_in[3];
  float*               out  = (float*)d_out;

  char* ws = (char*)d_ws;
  unsigned short* xbf  = (unsigned short*)ws; ws += (size_t)kROWS * kHID * 2;
  unsigned short* wt   = (unsigned short*)ws; ws += (size_t)kQKN * kHID * 2;
  unsigned short* qkb  = (unsigned short*)ws; ws += (size_t)kROWS * kQKN * 2;
  float*          vT   = (float*)ws;          ws += (size_t)kROWS * kNH * 4;
  float*          att  = (float*)ws;

  k_cast_bf16 <<<(kROWS * kHID) / 256, 256, 0, stream>>>(x, xbf, kROWS * kHID);
  k_w_transpose<<<(kQKN * kHID) / 256, 256, 0, stream>>>(Wqk, wt);
  k_gemm_qk   <<<dim3(kROWS / 16, kQKN / 128), 256, 0, stream>>>(xbf, wt, qkb);
  k_value     <<<kROWS / 16, 256, 0, stream>>>(x, Wv, vT);
  k_zero      <<<kROWS / 256, 256, 0, stream>>>(att, kROWS);
  k_attn      <<<dim3(kQS / 16, kNH, kBS), 256, 0, stream>>>(qkb, vT, mask, att);
  k_final     <<<kROWS / 256, 256, 0, stream>>>(att, mask, out);
}
